// NeuralDecisionTree_88888643158057
// MI455X (gfx1250) — matrix operand order repaired and checked by reading, not compiled, not run on hardware
//
#include <hip/hip_runtime.h>

typedef __attribute__((ext_vector_type(16))) _Float16 v16h;
typedef __attribute__((ext_vector_type(8)))  _Float16 v8h;
typedef __attribute__((ext_vector_type(8)))  float    v8f;
typedef __attribute__((ext_vector_type(4)))  float    v4f;

#define TREE_DEPTH 6
#define NLEAF 64
#define NINT  63
#define D_IN  512
#define O_OUT 256
#define ROWS_PER_WAVE 16
#define WAVES_PER_BLOCK 8
#define ROWS_PER_BLOCK (ROWS_PER_WAVE * WAVES_PER_BLOCK)

// ---------------- prep: row softmax of feature_selectors -> f16 W^T (K-contiguous) ----
// wW layout: [node 0..63][k 0..511], node 63 padded with zeros.
__global__ void ndt_prep_softmax(const float* __restrict__ fs, _Float16* __restrict__ wW) {
    const int node = blockIdx.x;      // 0..63
    const int t = threadIdx.x;        // 256 threads
    __shared__ float red[WAVES_PER_BLOCK];
    if (node >= NINT) {
        wW[node * D_IN + t] = (_Float16)0.0f;
        wW[node * D_IN + 256 + t] = (_Float16)0.0f;
        return;
    }
    const float* row = fs + node * D_IN;
    float v0 = row[t], v1 = row[t + 256];
    float m = fmaxf(v0, v1);
    for (int off = 16; off > 0; off >>= 1) m = fmaxf(m, __shfl_xor(m, off, 32));
    if ((t & 31) == 0) red[t >> 5] = m;
    __syncthreads();
    float bm = red[0];
    #pragma unroll
    for (int i = 1; i < WAVES_PER_BLOCK; ++i) bm = fmaxf(bm, red[i]);
    float e0 = __expf(v0 - bm), e1 = __expf(v1 - bm);
    float s = e0 + e1;
    for (int off = 16; off > 0; off >>= 1) s += __shfl_xor(s, off, 32);
    __syncthreads();
    if ((t & 31) == 0) red[t >> 5] = s;
    __syncthreads();
    float bs = 0.0f;
    #pragma unroll
    for (int i = 0; i < WAVES_PER_BLOCK; ++i) bs += red[i];
    float inv = 1.0f / bs;
    wW[node * D_IN + t]       = (_Float16)(e0 * inv);
    wW[node * D_IN + 256 + t] = (_Float16)(e1 * inv);
}

// ---------------- prep: convert leaf_embeddings (transposed, K-contiguous) + lin_w ----
__global__ void ndt_prep_convert(const float* __restrict__ leafE, const float* __restrict__ linw,
                                 _Float16* __restrict__ wLeaf, _Float16* __restrict__ wLin) {
    int i = blockIdx.x * blockDim.x + threadIdx.x;   // 0..65535
    if (i < NLEAF * O_OUT) {
        int n = i >> 6, k = i & 63;                  // wLeaf[n][k] = leafE[k][n]
        wLeaf[i] = (_Float16)leafE[k * O_OUT + n];
    }
    if (i < O_OUT * O_OUT) {
        wLin[i] = (_Float16)linw[i];                 // B[k][n'] = lin_w[n'][k] -> already K-contig
    }
}

// ---------------- fused main kernel: 8 waves/block, 16 rows/wave --------------------
__global__ __launch_bounds__(256, 1) void ndt_main(
    const float* __restrict__ x, const float* __restrict__ thr,
    const float* __restrict__ gamma, const float* __restrict__ beta,
    const float* __restrict__ linb,
    const _Float16* __restrict__ wW, const _Float16* __restrict__ wLeaf,
    const _Float16* __restrict__ wLin, float* __restrict__ out)
{
    extern __shared__ char lds[];
    const int lane = threadIdx.x & 31;
    const int wave = threadIdx.x >> 5;
    const int hh   = lane >> 4;     // half-wave: 0 or 1
    const int l16  = lane & 15;
    const int row0 = blockIdx.x * ROWS_PER_BLOCK + wave * ROWS_PER_WAVE;

    char* wbase = lds + wave * 8192;                  // 8 KB per-wave scratch
    float*    spl   = (float*)wbase;                  // phase A: splits [16][64] f32 (4 KB)
    _Float16* probs = (_Float16*)(wbase + 4096);      // phase A: probs  [16][64] f16 (2 KB)
    _Float16* outh  = (_Float16*)wbase;               // phase B: LN out [16][256] f16 (8 KB)

    // ---------------- GEMM1: feats = x @ softmax(W)^T   (M=16,N=64,K=512) ------------
    v8f acc1[4] = {};
    const float* xrow = x + (size_t)(row0 + l16) * D_IN;
    for (int ks = 0; ks < 16; ++ks) {
        const int k0 = ks * 32 + hh * 16;
        v4f xa = *(const v4f*)(xrow + k0);
        v4f xb = *(const v4f*)(xrow + k0 + 4);
        v4f xc = *(const v4f*)(xrow + k0 + 8);
        v4f xd = *(const v4f*)(xrow + k0 + 12);
        v16h a;
        #pragma unroll
        for (int i = 0; i < 4; ++i) {
            a[i]      = (_Float16)xa[i];
            a[4 + i]  = (_Float16)xb[i];
            a[8 + i]  = (_Float16)xc[i];
            a[12 + i] = (_Float16)xd[i];
        }
        const int kk = ks * 32 + hh * 16;
        #pragma unroll
        for (int nt = 0; nt < 4; ++nt) {
            v16h b = *(const v16h*)(wW + (nt * 16 + l16) * D_IN + kk);
            acc1[nt] = __builtin_amdgcn_wmma_f32_16x16x32_f16(
                false, a, false, b, (short)0, acc1[nt], false, false);
        }
    }

    // ---------------- sigmoid splits -> LDS ------------------------------------------
    #pragma unroll
    for (int nt = 0; nt < 4; ++nt) {
        const int n = nt * 16 + l16;
        const float th = (n < NINT) ? thr[n] : 0.0f;
        #pragma unroll
        for (int j = 0; j < 8; ++j) {
            const int m = j + hh * 8;
            float f = acc1[nt][j];
            float s = (n < NINT) ? (1.0f / (1.0f + __expf(-(f - th)))) : 0.0f;
            spl[m * 64 + n] = s;
        }
    }

    // ---------------- tree expansion: 16 rows x 64 leaves ----------------------------
    for (int it = 0; it < 32; ++it) {
        const int idx = it * 32 + lane;
        const int m = idx >> 6;
        const int leaf = idx & 63;
        float p = 1.0f;
        int node = 1;
        #pragma unroll
        for (int lev = 0; lev < TREE_DEPTH; ++lev) {
            const int bit = (leaf >> (5 - lev)) & 1;
            const float s = spl[m * 64 + (node - 1)];
            p *= bit ? s : (1.0f - s);
            node = 2 * node + bit;
        }
        probs[m * 64 + leaf] = (_Float16)p;
    }

    // ---------------- GEMM2: out = probs @ leafE   (M=16,N=256,K=64) -----------------
    v8f acc2[16] = {};
    #pragma unroll
    for (int ks = 0; ks < 2; ++ks) {
        const int koff = ks * 32 + hh * 16;
        v8h a0 = *(const v8h*)(probs + l16 * 64 + koff);
        v8h a1 = *(const v8h*)(probs + l16 * 64 + koff + 8);
        v16h a;
        #pragma unroll
        for (int i = 0; i < 8; ++i) { a[i] = a0[i]; a[8 + i] = a1[i]; }
        const int kk = ks * 32 + hh * 16;
        #pragma unroll
        for (int nt = 0; nt < 16; ++nt) {
            v16h b = *(const v16h*)(wLeaf + (nt * 16 + l16) * NLEAF + kk);
            acc2[nt] = __builtin_amdgcn_wmma_f32_16x16x32_f16(
                false, a, false, b, (short)0, acc2[nt], false, false);
        }
    }

    // ---------------- LayerNorm over O=256 (in-register, shfl reductions) ------------
    float gv[16], bv[16];
    #pragma unroll
    for (int nt = 0; nt < 16; ++nt) {
        const int n = nt * 16 + l16;
        gv[nt] = gamma[n];
        bv[nt] = beta[n];
    }
    #pragma unroll
    for (int j = 0; j < 8; ++j) {
        float s1 = 0.0f, s2 = 0.0f;
        #pragma unroll
        for (int nt = 0; nt < 16; ++nt) { float v = acc2[nt][j]; s1 += v; s2 += v * v; }
        #pragma unroll
        for (int off = 1; off < 16; off <<= 1) {
            s1 += __shfl_xor(s1, off, 32);
            s2 += __shfl_xor(s2, off, 32);
        }
        const float mean = s1 * (1.0f / 256.0f);
        const float var  = s2 * (1.0f / 256.0f) - mean * mean;
        const float rstd = rsqrtf(var + 1e-5f);
        const int m = j + hh * 8;
        #pragma unroll
        for (int nt = 0; nt < 16; ++nt) {
            const int n = nt * 16 + l16;
            float v = (acc2[nt][j] - mean) * rstd * gv[nt] + bv[nt];
            outh[m * O_OUT + n] = (_Float16)v;   // overlays spl/probs (already consumed)
        }
    }

    // ---------------- GEMM3: y = LNout @ lin_w^T   (M=16,N=256,K=256) ----------------
    v8f acc3[16] = {};
    for (int ks = 0; ks < 8; ++ks) {
        const int koff = ks * 32 + hh * 16;
        v8h a0 = *(const v8h*)(outh + l16 * O_OUT + koff);
        v8h a1 = *(const v8h*)(outh + l16 * O_OUT + koff + 8);
        v16h a;
        #pragma unroll
        for (int i = 0; i < 8; ++i) { a[i] = a0[i]; a[8 + i] = a1[i]; }
        const int kk = ks * 32 + hh * 16;
        #pragma unroll
        for (int nt = 0; nt < 16; ++nt) {
            v16h b = *(const v16h*)(wLin + (nt * 16 + l16) * O_OUT + kk);
            acc3[nt] = __builtin_amdgcn_wmma_f32_16x16x32_f16(
                false, a, false, b, (short)0, acc3[nt], false, false);
        }
    }

    // ---------------- epilogue: tanh(y + b) -> global --------------------------------
    #pragma unroll
    for (int nt = 0; nt < 16; ++nt) {
        const int n = nt * 16 + l16;
        const float bb = linb[n];
        #pragma unroll
        for (int j = 0; j < 8; ++j) {
            const int r = row0 + j + hh * 8;
            out[(size_t)r * O_OUT + n] = tanhf(acc3[nt][j] + bb);
        }
    }
}

extern "C" void kernel_launch(void* const* d_in, const int* in_sizes, int n_in,
                              void* d_out, int out_size, void* d_ws, size_t ws_size,
                              hipStream_t stream) {
    const float* x     = (const float*)d_in[0];
    const float* fs    = (const float*)d_in[1];
    const float* thr   = (const float*)d_in[2];
    const float* leafE = (const float*)d_in[3];
    const float* gamma = (const float*)d_in[4];
    const float* beta  = (const float*)d_in[5];
    const float* linw  = (const float*)d_in[6];
    const float* linb  = (const float*)d_in[7];
    float* out = (float*)d_out;

    // workspace: f16 weight copies (L2-resident), written fresh every call
    _Float16* wW    = (_Float16*)d_ws;                        // 64*512  = 64 KB
    _Float16* wLeaf = (_Float16*)((char*)d_ws + 64 * 1024);   // 256*64  = 32 KB
    _Float16* wLin  = (_Float16*)((char*)d_ws + 96 * 1024);   // 256*256 = 128 KB

    const int N = in_sizes[0] / D_IN;                         // 65536 rows
    ndt_prep_softmax<<<NLEAF, 256, 0, stream>>>(fs, wW);
    ndt_prep_convert<<<(O_OUT * O_OUT + 255) / 256, 256, 0, stream>>>(leafE, linw, wLeaf, wLin);
    ndt_main<<<N / ROWS_PER_BLOCK, 256, 64 * 1024, stream>>>(
        x, thr, gamma, beta, linb, wW, wLeaf, wLin, out);
}
